// MHA_52398601011392
// MI455X (gfx1250) — matrix operand order repaired and checked by reading, not compiled, not run on hardware
//
#include <hip/hip_runtime.h>
#include <hip/hip_bf16.h>
#include <math.h>

// ---------------- problem constants ----------------
#define BB   2
#define SS   2048
#define HH   2048
#define NHH  16
#define HDD  128
#define BS   (BB * SS)          // 4096
#define H3   (3 * HH)           // 6144

// ---------------- vector types ----------------
typedef __attribute__((ext_vector_type(16))) __bf16 v16bf;
typedef __attribute__((ext_vector_type(8)))  float  v8f;
typedef __attribute__((ext_vector_type(4)))  float  f32x4;
typedef __attribute__((ext_vector_type(8)))  unsigned short u16x8;

union frag16 {
    v16bf v;
    u16x8 h[2];
    unsigned short u[16];
};

__device__ __forceinline__ unsigned short f2bf_u(float f) {
    unsigned u = __builtin_bit_cast(unsigned, f);
    unsigned r = u + 0x7FFFu + ((u >> 16) & 1u);
    return (unsigned short)(r >> 16);
}

__device__ __forceinline__ v8f vzero8() {
    v8f z;
#pragma unroll
    for (int e = 0; e < 8; ++e) z[e] = 0.0f;
    return z;
}

__device__ __forceinline__ v8f wmma_bf16(v16bf a, v16bf b, v8f c) {
    return __builtin_amdgcn_wmma_f32_16x16x32_bf16(
        false, a, false, b, (short)0, c, false, false);
}

// ---- A-fragment (16x32, 16-bit): lane m=lane&15, K runs [8h,+8) and [16+8h,+8)
__device__ __forceinline__ v16bf load_afrag_bf16(const unsigned short* __restrict__ base,
                                                 int ld, int row0, int k0, int lane) {
    int m = lane & 15, hf = lane >> 4;
    const unsigned short* p = base + (size_t)(row0 + m) * ld + k0 + 8 * hf;
    frag16 r;
    r.h[0] = *(const u16x8*)(p);
    r.h[1] = *(const u16x8*)(p + 16);
    return r.v;
}

// ---- B-fragment (32x16, 16-bit): lane n=lane&15, K runs [8h,+8) and [16+8h,+8)
__device__ __forceinline__ v16bf load_bfrag_bf16(const unsigned short* __restrict__ bt,
                                                 int ld, int n0, int k0, int lane) {
    int n = lane & 15, hf = lane >> 4;
    const unsigned short* p = bt + (size_t)(n0 + n) * ld + k0 + 8 * hf;
    frag16 r;
    r.h[0] = *(const u16x8*)(p);
    r.h[1] = *(const u16x8*)(p + 16);
    return r.v;
}

// =====================================================================
// Kernel 0: bulk f32 -> bf16 (rounded) conversion, 8 elems/thread
// =====================================================================
__global__ __launch_bounds__(256) void cvt_bf16_kernel(
    const float* __restrict__ src, unsigned short* __restrict__ dst, int n8) {
    int i = blockIdx.x * blockDim.x + threadIdx.x;
    if (i >= n8) return;
    const f32x4* p = (const f32x4*)(src + (size_t)i * 8);
    f32x4 a = p[0], b = p[1];
    u16x8 o;
#pragma unroll
    for (int j = 0; j < 4; ++j) {
        o[j]     = f2bf_u(a[j]);
        o[4 + j] = f2bf_u(b[j]);
    }
    *(u16x8*)(dst + (size_t)i * 8) = o;
}

// =====================================================================
// GEMM: Y[M,N] = A[M,K]bf16 @ Bt[N,K]bf16^T + bias, f32 out
// one wave -> 32(M) x 64(N) tile, explicit ping-pong double buffering:
// two K-steps per iteration, loads for one buffer overlap WMMAs on other.
// =====================================================================
__global__ __launch_bounds__(256) void gemm_bf16_kernel(
    const unsigned short* __restrict__ A, const unsigned short* __restrict__ Bt,
    const float* __restrict__ bias, float* __restrict__ Y, int N) {
    const int lane = threadIdx.x & 31;
    const int wid  = blockIdx.x * (blockDim.x >> 5) + (threadIdx.x >> 5);
    const int NG = N / 64;
    const int mt = wid / NG, ng = wid % NG;
    const int m0 = mt * 32, n0 = ng * 64;

    v8f acc[2][4];
#pragma unroll
    for (int g = 0; g < 2; ++g)
#pragma unroll
        for (int t = 0; t < 4; ++t) acc[g][t] = vzero8();

    v16bf a0[2], b0[4], a1[2], b1[4];
    // prime buffer 0 with k=0
    a0[0] = load_afrag_bf16(A, HH, m0,      0, lane);
    a0[1] = load_afrag_bf16(A, HH, m0 + 16, 0, lane);
#pragma unroll
    for (int t = 0; t < 4; ++t) b0[t] = load_bfrag_bf16(Bt, HH, n0 + 16 * t, 0, lane);

    for (int k0 = 0; k0 < HH; k0 += 64) {
        // buffer 1 <- k0+32 (always in range: k0 <= HH-64)
        a1[0] = load_afrag_bf16(A, HH, m0,      k0 + 32, lane);
        a1[1] = load_afrag_bf16(A, HH, m0 + 16, k0 + 32, lane);
#pragma unroll
        for (int t = 0; t < 4; ++t)
            b1[t] = load_bfrag_bf16(Bt, HH, n0 + 16 * t, k0 + 32, lane);
        // compute on buffer 0 (k0) while buffer-1 loads are in flight
#pragma unroll
        for (int t = 0; t < 4; ++t) {
            acc[0][t] = wmma_bf16(a0[0], b0[t], acc[0][t]);
            acc[1][t] = wmma_bf16(a0[1], b0[t], acc[1][t]);
        }
        // buffer 0 <- k0+64 (skip on last iteration)
        if (k0 + 64 < HH) {
            a0[0] = load_afrag_bf16(A, HH, m0,      k0 + 64, lane);
            a0[1] = load_afrag_bf16(A, HH, m0 + 16, k0 + 64, lane);
#pragma unroll
            for (int t = 0; t < 4; ++t)
                b0[t] = load_bfrag_bf16(Bt, HH, n0 + 16 * t, k0 + 64, lane);
        }
        // compute on buffer 1 (k0+32) while buffer-0 loads are in flight
#pragma unroll
        for (int t = 0; t < 4; ++t) {
            acc[0][t] = wmma_bf16(a1[0], b1[t], acc[0][t]);
            acc[1][t] = wmma_bf16(a1[1], b1[t], acc[1][t]);
        }
    }

    const int nlo = lane & 15, hf = lane >> 4;
#pragma unroll
    for (int g = 0; g < 2; ++g)
#pragma unroll
        for (int t = 0; t < 4; ++t) {
            int n = n0 + 16 * t + nlo;
            float bs = bias[n];
#pragma unroll
            for (int r = 0; r < 8; ++r) {
                int m = m0 + 16 * g + r + 8 * hf;
                Y[(size_t)m * N + n] = acc[g][t][r] + bs;
            }
        }
}

// =====================================================================
// Kernel 2: RoPE on Q,K. qkv f32 [BS,6144] -> Qb,Kb bf16 [B,NH,S,HD]
// =====================================================================
__global__ __launch_bounds__(256) void rope_qk_kernel(
    const float* __restrict__ qkv, const float* __restrict__ cosT,
    const float* __restrict__ sinT, unsigned short* __restrict__ Qb,
    unsigned short* __restrict__ Kb) {
    int idx = blockIdx.x * blockDim.x + threadIdx.x;
    if (idx >= BB * SS * NHH * HDD) return;
    int d  = idx & (HDD - 1);
    int nh = (idx >> 7) & (NHH - 1);
    int s  = (idx >> 11) & (SS - 1);
    int b  = idx >> 22;

    size_t row = (size_t)(b * SS + s) * H3;
    int col = nh * HDD + d;
    int d2  = (d < HDD / 2) ? d + HDD / 2 : d - HDD / 2;
    float c  = cosT[s * HDD + d];
    float sn = sinT[s * HDD + d];

    float q  = qkv[row + col];
    float k  = qkv[row + HH + col];
    float qp = qkv[row + nh * HDD + d2];
    float kp = qkv[row + HH + nh * HDD + d2];
    float qr = (d < HDD / 2) ? -qp : qp;
    float kr = (d < HDD / 2) ? -kp : kp;

    size_t bh = (size_t)(b * NHH + nh);
    Qb[(bh * SS + s) * HDD + d] = f2bf_u(q * c + qr * sn);
    Kb[(bh * SS + s) * HDD + d] = f2bf_u(k * c + kr * sn);
}

// =====================================================================
// Kernel 2b: V transpose via LDS tile. qkv f32 -> Vt bf16 [B,NH,HD,S]
// =====================================================================
__global__ __launch_bounds__(256) void vtrans_kernel(
    const float* __restrict__ qkv, unsigned short* __restrict__ Vt) {
    __shared__ float tile[32][33];
    int t  = blockIdx.x;
    int dt = t & 3;
    int st = (t >> 2) & 63;
    int nh = (t >> 8) & 15;
    int b  = t >> 12;
    int tx = threadIdx.x & 31;
    int ty = threadIdx.x >> 5;
    int s0 = st * 32, d0 = dt * 32;

#pragma unroll
    for (int i = 0; i < 4; ++i) {
        int s = s0 + ty * 4 + i;
        tile[ty * 4 + i][tx] =
            qkv[(size_t)(b * SS + s) * H3 + 2 * HH + nh * HDD + d0 + tx];
    }
    __syncthreads();
#pragma unroll
    for (int i = 0; i < 4; ++i) {
        int d = d0 + ty * 4 + i;
        Vt[((size_t)(b * NHH + nh) * HDD + d) * SS + s0 + tx] =
            f2bf_u(tile[tx][ty * 4 + i]);
    }
}

// =====================================================================
// Kernel 3: causal flash attention. One wave per (b, nh, 16-row q tile).
// V fragments prefetched before softmax so their latency hides under exp.
// =====================================================================
__global__ __launch_bounds__(128) void flash_attn_kernel(
    const unsigned short* __restrict__ Qb, const unsigned short* __restrict__ Kb,
    const unsigned short* __restrict__ Vt, unsigned short* __restrict__ Obf) {
    __shared__ float          sc[4][16 * 32];
    __shared__ unsigned short pb[4][16 * 32];
    __shared__ float          alph[4][16];
    __shared__ float          linv[4][16];

    const int lane = threadIdx.x & 31;
    const int wv   = threadIdx.x >> 5;
    const int wid  = blockIdx.x * 4 + wv;          // 0..4095
    const int qt = wid & 127;
    const int nh = (wid >> 7) & 15;
    const int b  = wid >> 11;
    const int q0 = qt * 16;

    const size_t bh = (size_t)(b * NHH + nh);
    const unsigned short* Qp = Qb + bh * SS * HDD;
    const unsigned short* Kp = Kb + bh * SS * HDD;
    const unsigned short* Vp = Vt + bh * HDD * SS;

    v16bf qf[4];
#pragma unroll
    for (int c = 0; c < 4; ++c) qf[c] = load_afrag_bf16(Qp, HDD, q0, c * 32, lane);

    v8f o[8];
#pragma unroll
    for (int t = 0; t < 8; ++t) o[t] = vzero8();

    float m_i = -INFINITY, l_i = 0.0f;          // per-row state, lanes 0..15
    const float scale = 0.08838834764831845f;   // 1/sqrt(128)
    const int nlo = lane & 15, hf = lane >> 4;
    const int nblk = (q0 + 16 + 31) >> 5;

    for (int kb = 0; kb < nblk; ++kb) {
        const int key0 = kb * 32;
        v8f s0 = vzero8(), s1 = vzero8();
#pragma unroll
        for (int c = 0; c < 4; ++c) {
            v16bf k0f = load_bfrag_bf16(Kp, HDD, key0, c * 32, lane);
            v16bf k1f = load_bfrag_bf16(Kp, HDD, key0 + 16, c * 32, lane);
            s0 = wmma_bf16(qf[c], k0f, s0);
            s1 = wmma_bf16(qf[c], k1f, s1);
        }
        // prefetch V fragments now: independent of softmax, latency hides
        v16bf vf[8];
#pragma unroll
        for (int t = 0; t < 8; ++t)
            vf[t] = load_bfrag_bf16(Vp, SS, t * 16, key0, lane);

        // scores -> LDS (row = r + 8*hf, col = nlo / nlo+16)
#pragma unroll
        for (int r = 0; r < 8; ++r) {
            sc[wv][(r + 8 * hf) * 32 + nlo]      = s0[r] * scale;
            sc[wv][(r + 8 * hf) * 32 + 16 + nlo] = s1[r] * scale;
        }
        asm volatile("s_wait_dscnt 0" ::: "memory");
        __builtin_amdgcn_wave_barrier();

        // online softmax: lanes 0..15 each own one row
        if (lane < 16) {
            const int qg = q0 + lane;
            float mnew = m_i;
#pragma unroll 4
            for (int j = 0; j < 32; ++j) {
                float sv = sc[wv][lane * 32 + j];
                if ((key0 + j) <= qg && sv > mnew) mnew = sv;
            }
            float a = __expf(m_i - mnew);
            float ssum = 0.0f;
#pragma unroll 4
            for (int j = 0; j < 32; ++j) {
                float p = 0.0f;
                if ((key0 + j) <= qg)
                    p = __expf(sc[wv][lane * 32 + j] - mnew);
                pb[wv][lane * 32 + j] = f2bf_u(p);
                ssum += p;
            }
            l_i = l_i * a + ssum;
            m_i = mnew;
            alph[wv][lane] = a;
        }
        asm volatile("s_wait_dscnt 0" ::: "memory");
        __builtin_amdgcn_wave_barrier();

        // rescale running O by per-row alpha
        float ar[8];
#pragma unroll
        for (int r = 0; r < 8; ++r) ar[r] = alph[wv][r + 8 * hf];
#pragma unroll
        for (int t = 0; t < 8; ++t)
#pragma unroll
            for (int r = 0; r < 8; ++r) o[t][r] *= ar[r];

        // P as A-fragment from LDS
        frag16 pf;
        {
            const unsigned short* pp = &pb[wv][(size_t)nlo * 32 + 8 * hf];
            pf.h[0] = *(const u16x8*)(pp);
            pf.h[1] = *(const u16x8*)(pp + 16);
        }
        // O += P(16x32) @ V(32x128): 8 back-to-back WMMAs, V already in regs
#pragma unroll
        for (int t = 0; t < 8; ++t)
            o[t] = wmma_bf16(pf.v, vf[t], o[t]);
    }

    // finalize: divide by l, write bf16 [B,S, nh*HD+d]
    if (lane < 16) linv[wv][lane] = 1.0f / l_i;
    asm volatile("s_wait_dscnt 0" ::: "memory");
    __builtin_amdgcn_wave_barrier();
    float lr[8];
#pragma unroll
    for (int r = 0; r < 8; ++r) lr[r] = linv[wv][r + 8 * hf];
#pragma unroll
    for (int t = 0; t < 8; ++t)
#pragma unroll
        for (int r = 0; r < 8; ++r) {
            int q = q0 + r + 8 * hf;
            int dcol = t * 16 + nlo;
            Obf[(size_t)(b * SS + q) * HH + nh * HDD + dcol] = f2bf_u(o[t][r] * lr[r]);
        }
}

// =====================================================================
extern "C" void kernel_launch(void* const* d_in, const int* in_sizes, int n_in,
                              void* d_out, int out_size, void* d_ws, size_t ws_size,
                              hipStream_t stream) {
    const float* X    = (const float*)d_in[0];
    const float* cosT = (const float*)d_in[1];
    const float* sinT = (const float*)d_in[2];
    const float* Wqkv = (const float*)d_in[3];
    const float* bqkv = (const float*)d_in[4];
    const float* Wout = (const float*)d_in[5];
    const float* bout = (const float*)d_in[6];
    float* out = (float*)d_out;

    char* ws = (char*)d_ws;
    size_t off = 0;
    float* qkv = (float*)(ws + off);                     off += (size_t)BS * H3 * 4;  // 100.7 MB
    unsigned short* Xb    = (unsigned short*)(ws + off); off += (size_t)BS * HH * 2;  // 16.8 MB
    unsigned short* Wqkvb = (unsigned short*)(ws + off); off += (size_t)H3 * HH * 2;  // 25.2 MB
    unsigned short* Woutb = (unsigned short*)(ws + off); off += (size_t)HH * HH * 2;  //  8.4 MB
    unsigned short* Qb    = (unsigned short*)(ws + off); off += (size_t)BS * HH * 2;  // 16.8 MB
    unsigned short* Kb    = (unsigned short*)(ws + off); off += (size_t)BS * HH * 2;  // 16.8 MB
    unsigned short* Vt    = (unsigned short*)(ws + off); off += (size_t)BS * HH * 2;  // 16.8 MB
    unsigned short* Obf   = Xb;  // Xb dead after GEMM-1; reuse for attention out
    (void)ws_size; (void)in_sizes; (void)n_in; (void)out_size;

    // 0) one-time f32 -> bf16 conversions
    cvt_bf16_kernel<<<dim3((BS * HH / 8) / 256), dim3(256), 0, stream>>>(X, Xb, BS * HH / 8);
    cvt_bf16_kernel<<<dim3((H3 * HH / 8) / 256), dim3(256), 0, stream>>>(Wqkv, Wqkvb, H3 * HH / 8);
    cvt_bf16_kernel<<<dim3((HH * HH / 8) / 256), dim3(256), 0, stream>>>(Wout, Woutb, HH * HH / 8);

    // 1) QKV projection: 128 M-tiles x 96 N-groups = 12288 waves, 8 waves/block
    gemm_bf16_kernel<<<dim3((128 * 96) / 8), dim3(256), 0, stream>>>(Xb, Wqkvb, bqkv, qkv, H3);

    // 2) RoPE Q/K + V transpose to bf16
    rope_qk_kernel<<<dim3((BB * SS * NHH * HDD) / 256), dim3(256), 0, stream>>>(
        qkv, cosT, sinT, Qb, Kb);
    vtrans_kernel<<<dim3(BB * NHH * 64 * 4), dim3(256), 0, stream>>>(qkv, Vt);

    // 3) flash attention: 4096 waves, 4 waves/block
    flash_attn_kernel<<<dim3(4096 / 4), dim3(128), 0, stream>>>(Qb, Kb, Vt, Obf);

    // 4) output projection: 128 x 32 = 4096 waves, 8 waves/block
    gemm_bf16_kernel<<<dim3((128 * 32) / 8), dim3(256), 0, stream>>>(Obf, Woutb, bout, out, HH);
}
